// LFMMILoss_21208548508327
// MI455X (gfx1250) — matrix operand order repaired and checked by reading, not compiled, not run on hardware
//
#include <hip/hip_runtime.h>
#include <math.h>

// LF-MMI loss on gfx1250 (MI455X).
// Denominator DP  beta'[c] = logsumexp_j(beta[j]+trans[j,c]) + x_t[c]
// rewritten as an f16 WMMA GEMM via the max-shift trick:
//   p = exp(beta - rowmax),  E = exp(trans)  (time-invariant, register-resident)
//   beta' = rowmax + log(p @ E) + x_t
// One block = 16 batches (M=16), 16 waves = 16 N-tiles of 16 cols, K=256 in
// 8 x v_wmma_f32_16x16x32_f16 per wave per timestep, dual accumulator chains.

typedef __attribute__((ext_vector_type(16))) _Float16 v16h;
typedef __attribute__((ext_vector_type(8)))  float    v8f;

#define B_   64
#define T_   2000
#define C_   256
#define L_   128
#define MB   16      // batches (M rows) per den block
#define NWAVE 16     // waves per den block, one per 16-col N tile
#define NEGV (-1e30f)

__device__ __forceinline__ float logaddexpf_(float a, float b) {
    float mx = fmaxf(a, b);
    float mn = fminf(a, b);
    return mx + log1pf(expf(mn - mx));
}

// ---------------- numerator: linear label chain w/ self-loops ---------------
__global__ __launch_bounds__(L_) void lfmmi_num_kernel(
    const float* __restrict__ nn, const int* __restrict__ labels,
    const int* __restrict__ dur, float* __restrict__ ws_num)
{
    const int b = blockIdx.x;
    const int l = threadIdx.x;
    const int lab = labels[b * L_ + l];
    const float* __restrict__ xb = nn + (size_t)b * T_ * C_;
    const int d = dur[b];

    __shared__ float sh[L_];
    float alpha = (l == 0) ? xb[lab] : NEGV;

    for (int t = 1; t < T_; ++t) {
        sh[l] = alpha;
        __syncthreads();
        float shifted = (l > 0) ? sh[l - 1] : NEGV;
        float emit    = xb[(size_t)t * C_ + lab];
        float nv      = logaddexpf_(alpha, shifted) + emit;
        alpha = (t < d) ? nv : alpha;
        __syncthreads();
    }
    if (l == L_ - 1) ws_num[b] = alpha;
}

// ---------------- denominator: dense bigram DP via WMMA ---------------------
__global__ __launch_bounds__(32 * NWAVE) void lfmmi_den_kernel(
    const float* __restrict__ nn, const float* __restrict__ trans,
    const int* __restrict__ dur, float* __restrict__ ws_den)
{
    __shared__ float    beta_sh[MB * C_];   // 16 KB  running log-state
    __shared__ float    x_sh[MB * C_];      // 16 KB  staged x_t tile
    __shared__ _Float16 p_sh[MB * C_];      //  8 KB  exp(beta - max) in f16
    __shared__ float    m_sh[MB];
    __shared__ int      dur_sh[MB];

    const int tid   = threadIdx.x;
    const int w     = tid >> 5;        // wave id = N tile id = batch row id
    const int lane  = tid & 31;
    const int bbase = blockIdx.x * MB;

    // Register-resident B fragments: exp(trans) columns for this wave's tile.
    // B k order used here, same as A (32x16 f16): lanes 0-15 -> N=lane, K=0..15 ; lanes 16-31 -> K=16..31
    const int ncol  = w * 16 + (lane & 15);
    const int krow0 = (lane >> 4) * 16;
    v16h bfrag[8];
#pragma unroll
    for (int kf = 0; kf < 8; ++kf) {
#pragma unroll
        for (int i = 0; i < 16; ++i) {
            int k = kf * 32 + krow0 + i;
            bfrag[kf][i] = (_Float16)expf(trans[k * C_ + ncol]);
        }
    }

    // beta0 = x[0]; wave w owns batch row w of the block's 16-batch tile
    const float* __restrict__ rowp = nn + (size_t)(bbase + w) * T_ * C_;
#pragma unroll
    for (int i = 0; i < 8; ++i)
        beta_sh[w * C_ + lane + 32 * i] = rowp[lane + 32 * i];
    if (tid < MB) dur_sh[tid] = dur[bbase + tid];
    __syncthreads();

    // A k order used here, same as B (16x32 f16): lane<16 -> M=lane, K=0..15;
    // lane>=16 -> same M, K=16..31
    const int m_row_a = lane & 15;
    const int kh_a    = (lane >> 4) * 16;

    for (int t = 1; t < T_; ++t) {
        // ---- phase 1: per-row max + exp into f16, stage x_t (coalesced) ----
        float bv[8];
        float mx = NEGV;
#pragma unroll
        for (int i = 0; i < 8; ++i) {
            bv[i] = beta_sh[w * C_ + lane + 32 * i];
            mx = fmaxf(mx, bv[i]);
        }
#pragma unroll
        for (int off = 16; off > 0; off >>= 1)
            mx = fmaxf(mx, __shfl_xor(mx, off, 32));
#pragma unroll
        for (int i = 0; i < 8; ++i) {
            int j = lane + 32 * i;
            p_sh[w * C_ + j] = (_Float16)expf(bv[i] - mx);
            x_sh[w * C_ + j] = rowp[(size_t)t * C_ + j];
        }
        if (lane == 0) m_sh[w] = mx;
        if (t + 1 < T_)  // pull next timestep row toward L2/L0
            __builtin_prefetch(&rowp[(size_t)(t + 1) * C_ + lane * 8], 0, 1);
        __syncthreads();

        // ---- phase 2a: preload ALL A fragments so DS loads pipeline ----
        v16h afrag[8];
#pragma unroll
        for (int kf = 0; kf < 8; ++kf) {
            const int base0 = m_row_a * C_ + kf * 32 + kh_a;
#pragma unroll
            for (int i = 0; i < 8; ++i) {
                afrag[kf][i]     = p_sh[base0 + i];
                afrag[kf][8 + i] = p_sh[base0 + 8 + i];
            }
        }

        // ---- phase 2b: p(16x256) @ E(256x16), dual accumulator chains ----
        v8f acc0 = {};
        v8f acc1 = {};
#pragma unroll
        for (int kf = 0; kf < 8; kf += 2) {
            acc0 = __builtin_amdgcn_wmma_f32_16x16x32_f16(
                false, afrag[kf], false, bfrag[kf], (short)0, acc0, false, false);
            acc1 = __builtin_amdgcn_wmma_f32_16x16x32_f16(
                false, afrag[kf + 1], false, bfrag[kf + 1], (short)0, acc1, false, false);
        }

        // ---- phase 3: beta' = m + log(acc) + x_t, masked by duration ----
        // C/D layout: VGPR r, lane<16 -> (M=r, N=lane); lane>=16 -> (M=r+8, N=lane-16)
#pragma unroll
        for (int r = 0; r < 8; ++r) {
            int mrow = r + (lane >> 4) * 8;
            int col  = w * 16 + (lane & 15);
            float s  = acc0[r] + acc1[r];
            float nv = m_sh[mrow] + logf(s) + x_sh[mrow * C_ + col];
            if (t < dur_sh[mrow]) beta_sh[mrow * C_ + col] = nv;
        }
        __syncthreads();
    }

    // ---- final per-batch logsumexp over C states ----
    float bv[8];
    float mx = NEGV;
#pragma unroll
    for (int i = 0; i < 8; ++i) {
        bv[i] = beta_sh[w * C_ + lane + 32 * i];
        mx = fmaxf(mx, bv[i]);
    }
#pragma unroll
    for (int off = 16; off > 0; off >>= 1)
        mx = fmaxf(mx, __shfl_xor(mx, off, 32));
    float s = 0.f;
#pragma unroll
    for (int i = 0; i < 8; ++i) s += expf(bv[i] - mx);
#pragma unroll
    for (int off = 16; off > 0; off >>= 1)
        s += __shfl_xor(s, off, 32);
    if (lane == 0) ws_den[bbase + w] = mx + logf(s);
}

// ---------------- finalize: tot_score, tot_frames -------------------------
__global__ __launch_bounds__(B_) void lfmmi_finalize_kernel(
    const float* __restrict__ ws_num, const float* __restrict__ ws_den,
    const int* __restrict__ dur, float* __restrict__ out)
{
    __shared__ float rs[B_];
    __shared__ float rf[B_];
    const int i = threadIdx.x;
    rs[i] = ws_num[i] - ws_den[i];   // DEN_SCALE = 1.0
    rf[i] = (float)dur[i];
    __syncthreads();
    for (int off = B_ / 2; off > 0; off >>= 1) {
        if (i < off) { rs[i] += rs[i + off]; rf[i] += rf[i + off]; }
        __syncthreads();
    }
    if (i == 0) {
        out[0] = rs[0];
        out[1] = rf[0];
        out[2] = rf[0];
    }
}

extern "C" void kernel_launch(void* const* d_in, const int* in_sizes, int n_in,
                              void* d_out, int out_size, void* d_ws, size_t ws_size,
                              hipStream_t stream) {
    (void)in_sizes; (void)n_in; (void)out_size; (void)ws_size;
    const float* nn     = (const float*)d_in[0];   // (B,T,C) f32
    const float* trans  = (const float*)d_in[1];   // (C,C)   f32
    const int*   labels = (const int*)d_in[2];     // (B,L)   i32
    const int*   dur    = (const int*)d_in[3];     // (B,)    i32
    float* out    = (float*)d_out;                 // 3 floats
    float* ws_num = (float*)d_ws;                  // B floats
    float* ws_den = ws_num + B_;                   // B floats

    lfmmi_num_kernel<<<B_, L_, 0, stream>>>(nn, labels, dur, ws_num);
    lfmmi_den_kernel<<<B_ / MB, 32 * NWAVE, 0, stream>>>(nn, trans, dur, ws_den);
    lfmmi_finalize_kernel<<<1, B_, 0, stream>>>(ws_num, ws_den, dur, out);
}
